// SupervisedGraphUNetModel_56813827392003
// MI455X (gfx1250) — matrix operand order repaired and checked by reading, not compiled, not run on hardware
//
#include <hip/hip_runtime.h>
#include <stdint.h>

#define B_  64
#define N_  512
#define E_  256
#define H_  128
#define F_  9
#define V_  119

typedef __attribute__((ext_vector_type(16))) __bf16 v16bf;
typedef __attribute__((ext_vector_type(8)))  float  v8f;

// ---------------- workspace layout (bytes, per-graph arena) ----------------
#define ARENA     3462656ull
#define OFF_XEMB  0ull        // 512*256 bf16
#define OFF_A0    262144ull   // 512*512 bf16 (adj, diag 0)
#define OFF_AN0   786432ull   // 512*512 bf16 normalized
#define OFF_ANP1  1310720ull  // 256^2 bf16
#define OFF_ANP2  1441792ull  // 128^2 bf16
#define OFF_ANP3  1474560ull  // 64^2 bf16
#define OFF_AUG   1482752ull  // 512^2 bf16 scratch (reused per level)
#define OFF_AP1   2007040ull  // 256^2 bf16 pooled adj
#define OFF_AP2   2138112ull  // 128^2 bf16
#define OFF_AP3   2170880ull  // 64^2 bf16
#define OFF_DINV  2179072ull  // 512 f32 (reused per level)
#define OFF_XS0   2181120ull  // 512*128 bf16
#define OFF_XS1   2312192ull  // 256*128 bf16
#define OFF_XS2   2377728ull  // 128*128 bf16
#define OFF_XA    2410496ull  // 512*128 bf16 generic
#define OFF_XB    2541568ull  // 512*128 bf16 generic
#define OFF_XWT   2672640ull  // up to 256*512 bf16 (xW transposed)
#define OFF_PERM0 2934784ull
#define OFF_PERM1 2935808ull
#define OFF_PERM2 2936320ull
#define OFF_VALS0 2936576ull
#define OFF_VALS1 2937600ull
#define OFF_VALS2 2938112ull
#define OFF_HNODE 2938368ull  // 512*256 f32
// globals past the 64 arenas
#define GOFF      (64ull*ARENA)
#define OFF_W0T   (GOFF + 0ull)       // 128x256 bf16
#define OFF_WDT   (GOFF + 65536ull)   // 3 x 128x128 bf16
#define OFF_WUT   (GOFF + 163840ull)  // 2 x 128x128 bf16
#define OFF_WLT   (GOFF + 229376ull)  // 256x128 bf16
#define OFF_HG    (GOFF + 294912ull)  // 64x256 f32
#define OFF_H1    (GOFF + 360448ull)  // 64x256 f32
#define OFF_H2    (GOFF + 425984ull)  // 64x128 f32

__device__ __forceinline__ float bf2f(uint16_t h) {
  return __uint_as_float(((uint32_t)h) << 16);
}
__device__ __forceinline__ uint16_t f2bf(float f) {
  uint32_t x = __float_as_uint(f);
  x += 0x7fffu + ((x >> 16) & 1u);   // round-to-nearest-even
  return (uint16_t)(x >> 16);
}

// ---------------- batched bf16 WMMA GEMM: C[M,N] = A[M,K] * B[K,N] ---------
// A row-major; B supplied TRANSPOSED (BT is [N,K] row-major).
// Each wave owns a 32x32 C tile (2x2 accumulators): 2 A-frags x 2 B-frags
// feed 4 v_wmma per K-step -> 2 b128 loads per wmma (register-level reuse).
// Block = 8 waves (2x4) -> 64x128 block tile. M%64==0, N%128==0, K%32==0.
// DIAG1: treat operand diagonals as 1.0 (A1 = A + I); ZERODIAG: zero C diag.
// OUT_TRANS: store C transposed ([N,M], ldc = M). BIAS: + bias[col] (f32).
template<bool DIAG1, bool ZERODIAG, bool OUT_TRANS, bool OUT_F32, bool DORELU, bool BIAS>
__global__ __launch_bounds__(256)
void gemm_bf16_wmma(const char* Abase, size_t aStride, int lda,
                    const char* Bbase, size_t bStride, int ldb,
                    char* Cbase, size_t cStride, int ldc,
                    const float* bias, int K)
{
  const int b = blockIdx.z;
  const uint16_t* A  = (const uint16_t*)(Abase + (size_t)b * aStride);
  const uint16_t* BT = (const uint16_t*)(Bbase + (size_t)b * bStride);
  const int lane = threadIdx.x & 31;
  const int wave = threadIdx.x >> 5;
  const int tileM = blockIdx.y * 64  + (wave >> 2) * 32;
  const int tileN = blockIdx.x * 128 + (wave & 3) * 32;
  const int hi = lane >> 4;     // lane group
  const int lx = lane & 15;

  union Vec { v16bf v; uint4 q[2]; uint16_t s[16]; };
  v8f acc00 = {}, acc01 = {}, acc10 = {}, acc11 = {};

  const int rA0 = tileM + lx, rA1 = tileM + 16 + lx;   // A rows this lane loads
  const int rB0 = tileN + lx, rB1 = tileN + 16 + lx;   // BT rows (= C columns)
  const uint16_t* aRow0 = A  + (size_t)rA0 * lda;
  const uint16_t* aRow1 = A  + (size_t)rA1 * lda;
  const uint16_t* bRow0 = BT + (size_t)rB0 * ldb;
  const uint16_t* bRow1 = BT + (size_t)rB1 * ldb;

  for (int k0 = 0; k0 < K; k0 += 32) {
    Vec a0, a1, b0, b1;
    // A 16-bit layout: elem i<8 -> K=k0+hi*8+i ; elem i>=8 -> K=k0+16+hi*8+(i-8)
    a0.q[0] = *(const uint4*)(aRow0 + k0 + hi * 8);
    a0.q[1] = *(const uint4*)(aRow0 + k0 + 16 + hi * 8);
    a1.q[0] = *(const uint4*)(aRow1 + k0 + hi * 8);
    a1.q[1] = *(const uint4*)(aRow1 + k0 + 16 + hi * 8);
    // B 16-bit layout: elem i<8 -> K=k0+hi*8+i ; elem i>=8 -> K=k0+16+hi*8+(i-8) (column = BT row)
    b0.q[0] = *(const uint4*)(bRow0 + k0 + hi * 8);
    b0.q[1] = *(const uint4*)(bRow0 + k0 + 16 + hi * 8);
    b1.q[0] = *(const uint4*)(bRow1 + k0 + hi * 8);
    b1.q[1] = *(const uint4*)(bRow1 + k0 + 16 + hi * 8);
    if (DIAG1) {
      int d;
      d = rA0 - (k0 + hi * 8);        if (d >= 0 && d < 8)  a0.s[d]     = 0x3F80;
      d = rA0 - (k0 + 16 + hi * 8);   if (d >= 0 && d < 8)  a0.s[8 + d] = 0x3F80;
      d = rA1 - (k0 + hi * 8);        if (d >= 0 && d < 8)  a1.s[d]     = 0x3F80;
      d = rA1 - (k0 + 16 + hi * 8);   if (d >= 0 && d < 8)  a1.s[8 + d] = 0x3F80;
      d = rB0 - (k0 + hi * 8);        if (d >= 0 && d < 8)  b0.s[d]     = 0x3F80;
      d = rB0 - (k0 + 16 + hi * 8);   if (d >= 0 && d < 8)  b0.s[8 + d] = 0x3F80;
      d = rB1 - (k0 + hi * 8);        if (d >= 0 && d < 8)  b1.s[d]     = 0x3F80;
      d = rB1 - (k0 + 16 + hi * 8);   if (d >= 0 && d < 8)  b1.s[8 + d] = 0x3F80;
    }
    acc00 = __builtin_amdgcn_wmma_f32_16x16x32_bf16(false, a0.v, false, b0.v, (short)0, acc00, false, false);
    acc01 = __builtin_amdgcn_wmma_f32_16x16x32_bf16(false, a0.v, false, b1.v, (short)0, acc01, false, false);
    acc10 = __builtin_amdgcn_wmma_f32_16x16x32_bf16(false, a1.v, false, b0.v, (short)0, acc10, false, false);
    acc11 = __builtin_amdgcn_wmma_f32_16x16x32_bf16(false, a1.v, false, b1.v, (short)0, acc11, false, false);
  }

  char* Cb = Cbase + (size_t)b * cStride;
  auto store_tile = [&](const v8f& acc, int moff, int noff) {
    const int col = tileN + noff + lx;
    const float bv = BIAS ? bias[col] : 0.0f;
#pragma unroll
    for (int r = 0; r < 8; ++r) {
      int m = tileM + moff + hi * 8 + r;   // C layout: lane = n + 16*(mrel>>3)
      float v = acc[r] + bv;
      if (DORELU) v = v > 0.0f ? v : 0.0f;
      if (ZERODIAG && m == col) v = 0.0f;
      if (OUT_F32) {
        float* C = (float*)Cb;
        if (OUT_TRANS) C[(size_t)col * ldc + m] = v;
        else           C[(size_t)m * ldc + col] = v;
      } else {
        uint16_t* C = (uint16_t*)Cb;
        uint16_t hv = f2bf(v);
        if (OUT_TRANS) C[(size_t)col * ldc + m] = hv;
        else           C[(size_t)m * ldc + col] = hv;
      }
    }
  };
  store_tile(acc00, 0, 0);
  store_tile(acc01, 0, 16);
  store_tile(acc10, 16, 0);
  store_tile(acc11, 16, 16);
}

// ---------------- small kernels -------------------------------------------
__global__ void prep_weights_kernel(const float* W0, const float* Wd, const float* Wu,
                                    const float* Wlast, char* ws)
{
  uint16_t* W0T = (uint16_t*)(ws + OFF_W0T);
  uint16_t* WDT = (uint16_t*)(ws + OFF_WDT);
  uint16_t* WUT = (uint16_t*)(ws + OFF_WUT);
  uint16_t* WLT = (uint16_t*)(ws + OFF_WLT);
  for (int t = blockIdx.x * 256 + threadIdx.x; t < 147456; t += gridDim.x * 256) {
    if (t < 32768) {                       // W0 (E,H) -> W0T [H,E]
      int nn = t / 256, kk = t % 256;
      W0T[t] = f2bf(W0[(size_t)kk * H_ + nn]);
    } else if (t < 81920) {                // Wd (3,H,H) -> WdT [H,H]
      int u = t - 32768; int d = u / 16384; u %= 16384;
      int nn = u / H_, kk = u % H_;
      WDT[d * 16384 + nn * H_ + kk] = f2bf(Wd[((size_t)d * H_ + kk) * H_ + nn]);
    } else if (t < 114688) {               // Wu (2,H,H) -> WuT
      int u = t - 81920; int d = u / 16384; u %= 16384;
      int nn = u / H_, kk = u % H_;
      WUT[d * 16384 + nn * H_ + kk] = f2bf(Wu[((size_t)d * H_ + kk) * H_ + nn]);
    } else {                               // Wlast (H,E) -> WlastT [E,H]
      int u = t - 114688;
      int nn = u / H_, kk = u % H_;
      WLT[nn * H_ + kk] = f2bf(Wlast[(size_t)kk * E_ + nn]);
    }
  }
}

__global__ void atom_encoder_kernel(const int* xa, const float* emb, char* ws)
{
  int t = blockIdx.x * 256 + threadIdx.x;
  if (t >= B_ * N_ * E_) return;
  int e = t % E_;
  int bn = t / E_;
  const int* feat = xa + (size_t)bn * F_;
  float s = 0.0f;
#pragma unroll
  for (int f = 0; f < F_; ++f)
    s += emb[((size_t)f * V_ + feat[f]) * E_ + e];
  int b = bn / N_, n = bn % N_;
  ((uint16_t*)(ws + (size_t)b * ARENA + OFF_XEMB))[(size_t)n * E_ + e] = f2bf(s);
}

__global__ void deg_f32_kernel(const float* adj, char* ws)
{
  int t = blockIdx.x * 256 + threadIdx.x;
  if (t >= B_ * N_) return;
  int b = t / N_, i = t % N_;
  const float* row = adj + ((size_t)b * N_ + i) * N_;
  float s = 2.0f;
  for (int j = 0; j < N_; ++j) s += row[j];
  ((float*)(ws + (size_t)b * ARENA + OFF_DINV))[i] = rsqrtf(s);
}

__global__ void norm0_kernel(const float* adj, char* ws)
{
  long long t = (long long)blockIdx.x * 256 + threadIdx.x;
  if (t >= (long long)B_ * N_ * N_) return;
  int b = (int)(t / (N_ * N_));
  int u = (int)(t % (N_ * N_));
  int r = u / N_, c = u % N_;
  char* ar = ws + (size_t)b * ARENA;
  float a = adj[(size_t)b * N_ * N_ + u];
  const float* dinv = (const float*)(ar + OFF_DINV);
  ((uint16_t*)(ar + OFF_AN0))[u] = f2bf(dinv[r] * (a + (r == c ? 2.0f : 0.0f)) * dinv[c]);
  ((uint16_t*)(ar + OFF_A0))[u]  = f2bf(a);
}

__global__ void deg_bf_kernel(char* ws, size_t aOff, int n)
{
  int t = blockIdx.x * 256 + threadIdx.x;
  if (t >= B_ * n) return;
  int b = t / n, i = t % n;
  char* ar = ws + (size_t)b * ARENA;
  const uint16_t* row = (const uint16_t*)(ar + aOff) + (size_t)i * n;
  float s = 2.0f;
  for (int j = 0; j < n; ++j) s += bf2f(row[j]);
  ((float*)(ar + OFF_DINV))[i] = rsqrtf(s);
}

__global__ void norm_bf_kernel(char* ws, size_t aOff, size_t dstOff, int n)
{
  int t = blockIdx.x * 256 + threadIdx.x;
  if (t >= B_ * n * n) return;
  int b = t / (n * n);
  int u = t % (n * n);
  int r = u / n, c = u % n;
  char* ar = ws + (size_t)b * ARENA;
  float a = bf2f(((const uint16_t*)(ar + aOff))[u]);
  const float* dinv = (const float*)(ar + OFF_DINV);
  ((uint16_t*)(ar + dstOff))[u] = f2bf(dinv[r] * (a + (r == c ? 2.0f : 0.0f)) * dinv[c]);
}

__global__ void topk_kernel(char* ws, size_t xOff, const float* pw, int n, int k,
                            size_t permOff, size_t valsOff)
{
  __shared__ float ss[512];
  __shared__ int   si[512];
  int b = blockIdx.x, tid = threadIdx.x;
  char* ar = ws + (size_t)b * ARENA;
  const uint16_t* x = (const uint16_t*)(ar + xOff);
  float wn = 0.0f;
  for (int h = 0; h < H_; ++h) { float w = pw[h]; wn += w * w; }
  wn = sqrtf(wn);
  const uint16_t* xr = x + (size_t)tid * H_;
  float dot = 0.0f;
  for (int h = 0; h < H_; ++h) dot += bf2f(xr[h]) * pw[h];
  ss[tid] = tanhf(dot / wn);
  si[tid] = tid;
  // bitonic sort: descending score, ascending index on ties
  for (int size = 2; size <= n; size <<= 1)
    for (int stride = size >> 1; stride > 0; stride >>= 1) {
      __syncthreads();
      int p = tid ^ stride;
      if (p > tid) {
        float s1 = ss[tid], s2 = ss[p];
        int   i1 = si[tid], i2 = si[p];
        bool pre = (s1 > s2) || (s1 == s2 && i1 < i2);
        bool up  = ((tid & size) == 0);
        if (up ? !pre : pre) { ss[tid] = s2; ss[p] = s1; si[tid] = i2; si[p] = i1; }
      }
    }
  __syncthreads();
  if (tid < k) {
    ((int*)  (ar + permOff))[tid] = si[tid];
    ((float*)(ar + valsOff))[tid] = ss[tid];
  }
}

__global__ void gatherA_kernel(char* ws, size_t augOff, size_t permOff, size_t dstOff,
                               int n, int k)
{
  int t = blockIdx.x * 256 + threadIdx.x;
  if (t >= B_ * k * k) return;
  int b = t / (k * k);
  int u = t % (k * k);
  int i = u / k, j = u % k;
  char* ar = ws + (size_t)b * ARENA;
  const int* perm = (const int*)(ar + permOff);
  ((uint16_t*)(ar + dstOff))[u] =
      ((const uint16_t*)(ar + augOff))[(size_t)perm[i] * n + perm[j]];
}

__global__ void gatherX_kernel(char* ws, size_t xOff, size_t permOff, size_t valsOff,
                               size_t dstOff, int k)
{
  int t = blockIdx.x * 256 + threadIdx.x;
  if (t >= B_ * k * H_) return;
  int b = t / (k * H_);
  int u = t % (k * H_);
  int i = u / H_, h = u % H_;
  char* ar = ws + (size_t)b * ARENA;
  const int*   perm = (const int*)(ar + permOff);
  const float* vals = (const float*)(ar + valsOff);
  float v = bf2f(((const uint16_t*)(ar + xOff))[(size_t)perm[i] * H_ + h]) * vals[i];
  ((uint16_t*)(ar + dstOff))[u] = f2bf(v);
}

__global__ void copy_bf_kernel(char* ws, size_t dstOff, size_t srcOff, int cnt)
{
  int t = blockIdx.x * 256 + threadIdx.x;
  if (t >= B_ * cnt) return;
  int b = t / cnt, u = t % cnt;
  char* ar = ws + (size_t)b * ARENA;
  ((uint16_t*)(ar + dstOff))[u] = ((const uint16_t*)(ar + srcOff))[u];
}

__global__ void scatter_add_kernel(char* ws, size_t dstOff, size_t srcOff,
                                   size_t permOff, int k)
{
  int t = blockIdx.x * 256 + threadIdx.x;
  if (t >= B_ * k * H_) return;
  int b = t / (k * H_);
  int u = t % (k * H_);
  int i = u / H_, h = u % H_;
  char* ar = ws + (size_t)b * ARENA;
  const int* perm = (const int*)(ar + permOff);
  uint16_t* dst = (uint16_t*)(ar + dstOff);
  size_t tg = (size_t)perm[i] * H_ + h;
  float v = bf2f(dst[tg]) + bf2f(((const uint16_t*)(ar + srcOff))[u]);
  dst[tg] = f2bf(v);
}

__global__ void pool_kernel(char* ws)
{
  int t = blockIdx.x * 256 + threadIdx.x;
  if (t >= B_ * E_) return;
  int b = t / E_, e = t % E_;
  const float* hn = (const float*)(ws + (size_t)b * ARENA + OFF_HNODE);
  float s = 0.0f;
  for (int n = 0; n < N_; ++n) s += hn[(size_t)n * E_ + e];
  ((float*)(ws + OFF_HG))[t] = s;
}

__global__ void mlp_gemm_kernel(char* ws, size_t inOff, size_t outOff,
                                const float* W, const float* bias, int Kc, int Nc)
{
  int t = blockIdx.x * 256 + threadIdx.x;
  if (t >= B_ * Nc) return;
  int r = t / Nc, c = t % Nc;
  const float* in = (const float*)(ws + inOff) + (size_t)r * Kc;
  float s = bias[c];
  for (int k = 0; k < Kc; ++k) s += in[k] * W[(size_t)k * Nc + c];
  ((float*)(ws + outOff))[t] = s;
}

__global__ void bn_kernel(char* ws, size_t off, const float* g, const float* be, int Nc)
{
  int c = blockIdx.x * blockDim.x + threadIdx.x;
  if (c >= Nc) return;
  float* h = (float*)(ws + off);
  float m = 0.0f;
  for (int r = 0; r < B_; ++r) m += h[(size_t)r * Nc + c];
  m /= (float)B_;
  float v = 0.0f;
  for (int r = 0; r < B_; ++r) { float d = h[(size_t)r * Nc + c] - m; v += d * d; }
  v /= (float)B_;
  float inv = rsqrtf(v + 1e-5f);
  for (int r = 0; r < B_; ++r)
    h[(size_t)r * Nc + c] = (h[(size_t)r * Nc + c] - m) * inv * g[c] + be[c];
}

__global__ void mlp_out_kernel(char* ws, const float* P3w, const float* P3b, float* out)
{
  int r = threadIdx.x;
  if (r >= B_) return;
  const float* h = (const float*)(ws + OFF_H2) + (size_t)r * H_;
  float s = P3b[0];
  for (int k = 0; k < H_; ++k) s += h[k] * P3w[k];
  out[r] = s;
}

// ---------------- driver ---------------------------------------------------
#define GEMM(D1, ZD, TR, F32, RL, BI, M, Nn, K, Ab, aS, la, Bb, bS, lb, Cb, cS, lc, bias) \
  gemm_bf16_wmma<D1, ZD, TR, F32, RL, BI>                                                 \
      <<<dim3((Nn) / 128, (M) / 64, B_), 256, 0, stream>>>(                               \
          (const char*)(Ab), (size_t)(aS), (la), (const char*)(Bb), (size_t)(bS), (lb),   \
          (char*)(Cb), (size_t)(cS), (lc), (bias), (K))

extern "C" void kernel_launch(void* const* d_in, const int* in_sizes, int n_in,
                              void* d_out, int out_size, void* d_ws, size_t ws_size,
                              hipStream_t stream)
{
  (void)in_sizes; (void)n_in; (void)out_size; (void)ws_size;
  const int*   x_atoms = (const int*)d_in[0];
  const float* adj     = (const float*)d_in[1];
  const float* emb     = (const float*)d_in[2];
  const float* W0      = (const float*)d_in[3];
  const float* b0      = (const float*)d_in[4];
  const float* Wd      = (const float*)d_in[5];
  const float* bd      = (const float*)d_in[6];
  const float* pw      = (const float*)d_in[7];
  const float* Wu      = (const float*)d_in[8];
  const float* bu      = (const float*)d_in[9];
  const float* Wlast   = (const float*)d_in[10];
  const float* blast   = (const float*)d_in[11];
  const float* P1w     = (const float*)d_in[12];
  const float* P1b     = (const float*)d_in[13];
  const float* g1      = (const float*)d_in[14];
  const float* be1     = (const float*)d_in[15];
  const float* P2w     = (const float*)d_in[16];
  const float* P2b     = (const float*)d_in[17];
  const float* g2      = (const float*)d_in[18];
  const float* be2     = (const float*)d_in[19];
  const float* P3w     = (const float*)d_in[20];
  const float* P3b     = (const float*)d_in[21];
  char* ws = (char*)d_ws;
  auto eg = [](long long total) { return dim3((unsigned)((total + 255) / 256)); };

  prep_weights_kernel<<<64, 256, 0, stream>>>(W0, Wd, Wu, Wlast, ws);
  atom_encoder_kernel<<<eg((long long)B_ * N_ * E_), 256, 0, stream>>>(x_atoms, emb, ws);
  deg_f32_kernel<<<eg(B_ * N_), 256, 0, stream>>>(adj, ws);
  norm0_kernel<<<eg((long long)B_ * N_ * N_), 256, 0, stream>>>(adj, ws);

  // GCN0: x = relu(An0 @ (xemb @ W0) + b0)
  GEMM(false, false, true,  false, false, false, 512, 128, 256,
       ws + OFF_XEMB, ARENA, 256, ws + OFF_W0T, 0, 256, ws + OFF_XWT, ARENA, 512, (const float*)nullptr);
  GEMM(false, false, false, false, true,  true,  512, 128, 512,
       ws + OFF_AN0, ARENA, 512, ws + OFF_XWT, ARENA, 512, ws + OFF_XS0, ARENA, 128, b0);

  // ---- level 0 (n=512 -> k=256) ----
  GEMM(true, true, false, false, false, false, 512, 512, 512,
       ws + OFF_A0, ARENA, 512, ws + OFF_A0, ARENA, 512, ws + OFF_AUG, ARENA, 512, (const float*)nullptr);
  topk_kernel<<<B_, 512, 0, stream>>>(ws, OFF_XS0, pw + 0, 512, 256, OFF_PERM0, OFF_VALS0);
  gatherA_kernel<<<eg((long long)B_ * 256 * 256), 256, 0, stream>>>(ws, OFF_AUG, OFF_PERM0, OFF_AP1, 512, 256);
  gatherX_kernel<<<eg((long long)B_ * 256 * H_), 256, 0, stream>>>(ws, OFF_XS0, OFF_PERM0, OFF_VALS0, OFF_XA, 256);
  deg_bf_kernel<<<eg(B_ * 256), 256, 0, stream>>>(ws, OFF_AP1, 256);
  norm_bf_kernel<<<eg((long long)B_ * 256 * 256), 256, 0, stream>>>(ws, OFF_AP1, OFF_ANP1, 256);
  GEMM(false, false, true,  false, false, false, 256, 128, 128,
       ws + OFF_XA, ARENA, 128, ws + OFF_WDT, 0, 128, ws + OFF_XWT, ARENA, 256, (const float*)nullptr);
  GEMM(false, false, false, false, true,  true,  256, 128, 256,
       ws + OFF_ANP1, ARENA, 256, ws + OFF_XWT, ARENA, 256, ws + OFF_XS1, ARENA, 128, bd + 0);

  // ---- level 1 (n=256 -> k=128) ----
  GEMM(true, true, false, false, false, false, 256, 256, 256,
       ws + OFF_AP1, ARENA, 256, ws + OFF_AP1, ARENA, 256, ws + OFF_AUG, ARENA, 256, (const float*)nullptr);
  topk_kernel<<<B_, 256, 0, stream>>>(ws, OFF_XS1, pw + 128, 256, 128, OFF_PERM1, OFF_VALS1);
  gatherA_kernel<<<eg((long long)B_ * 128 * 128), 256, 0, stream>>>(ws, OFF_AUG, OFF_PERM1, OFF_AP2, 256, 128);
  gatherX_kernel<<<eg((long long)B_ * 128 * H_), 256, 0, stream>>>(ws, OFF_XS1, OFF_PERM1, OFF_VALS1, OFF_XA, 128);
  deg_bf_kernel<<<eg(B_ * 128), 256, 0, stream>>>(ws, OFF_AP2, 128);
  norm_bf_kernel<<<eg((long long)B_ * 128 * 128), 256, 0, stream>>>(ws, OFF_AP2, OFF_ANP2, 128);
  GEMM(false, false, true,  false, false, false, 128, 128, 128,
       ws + OFF_XA, ARENA, 128, ws + OFF_WDT + 32768, 0, 128, ws + OFF_XWT, ARENA, 128, (const float*)nullptr);
  GEMM(false, false, false, false, true,  true,  128, 128, 128,
       ws + OFF_ANP2, ARENA, 128, ws + OFF_XWT, ARENA, 128, ws + OFF_XS2, ARENA, 128, bd + 128);

  // ---- level 2 (n=128 -> k=64) ----
  GEMM(true, true, false, false, false, false, 128, 128, 128,
       ws + OFF_AP2, ARENA, 128, ws + OFF_AP2, ARENA, 128, ws + OFF_AUG, ARENA, 128, (const float*)nullptr);
  topk_kernel<<<B_, 128, 0, stream>>>(ws, OFF_XS2, pw + 256, 128, 64, OFF_PERM2, OFF_VALS2);
  gatherA_kernel<<<eg((long long)B_ * 64 * 64), 256, 0, stream>>>(ws, OFF_AUG, OFF_PERM2, OFF_AP3, 128, 64);
  gatherX_kernel<<<eg((long long)B_ * 64 * H_), 256, 0, stream>>>(ws, OFF_XS2, OFF_PERM2, OFF_VALS2, OFF_XA, 64);
  deg_bf_kernel<<<eg(B_ * 64), 256, 0, stream>>>(ws, OFF_AP3, 64);
  norm_bf_kernel<<<eg((long long)B_ * 64 * 64), 256, 0, stream>>>(ws, OFF_AP3, OFF_ANP3, 64);
  GEMM(false, false, true,  false, false, false, 64, 128, 128,
       ws + OFF_XA, ARENA, 128, ws + OFF_WDT + 65536, 0, 128, ws + OFF_XWT, ARENA, 64, (const float*)nullptr);
  GEMM(false, false, false, false, true,  true,  64, 128, 64,
       ws + OFF_ANP3, ARENA, 64, ws + OFF_XWT, ARENA, 64, ws + OFF_XB, ARENA, 128, bd + 256);

  // ---- up path i=0 (j=2, 128 nodes) ----
  copy_bf_kernel<<<eg((long long)B_ * 128 * H_), 256, 0, stream>>>(ws, OFF_XA, OFF_XS2, 128 * H_);
  scatter_add_kernel<<<eg((long long)B_ * 64 * H_), 256, 0, stream>>>(ws, OFF_XA, OFF_XB, OFF_PERM2, 64);
  GEMM(false, false, true,  false, false, false, 128, 128, 128,
       ws + OFF_XA, ARENA, 128, ws + OFF_WUT, 0, 128, ws + OFF_XWT, ARENA, 128, (const float*)nullptr);
  GEMM(false, false, false, false, true,  true,  128, 128, 128,
       ws + OFF_ANP2, ARENA, 128, ws + OFF_XWT, ARENA, 128, ws + OFF_XB, ARENA, 128, bu + 0);

  // ---- up path i=1 (j=1, 256 nodes) ----
  copy_bf_kernel<<<eg((long long)B_ * 256 * H_), 256, 0, stream>>>(ws, OFF_XA, OFF_XS1, 256 * H_);
  scatter_add_kernel<<<eg((long long)B_ * 128 * H_), 256, 0, stream>>>(ws, OFF_XA, OFF_XB, OFF_PERM1, 128);
  GEMM(false, false, true,  false, false, false, 256, 128, 128,
       ws + OFF_XA, ARENA, 128, ws + OFF_WUT + 32768, 0, 128, ws + OFF_XWT, ARENA, 256, (const float*)nullptr);
  GEMM(false, false, false, false, true,  true,  256, 128, 256,
       ws + OFF_ANP1, ARENA, 256, ws + OFF_XWT, ARENA, 256, ws + OFF_XB, ARENA, 128, bu + 128);

  // ---- up path i=2 (j=0, 512 nodes, Wlast, f32 out) ----
  copy_bf_kernel<<<eg((long long)B_ * 512 * H_), 256, 0, stream>>>(ws, OFF_XA, OFF_XS0, 512 * H_);
  scatter_add_kernel<<<eg((long long)B_ * 256 * H_), 256, 0, stream>>>(ws, OFF_XA, OFF_XB, OFF_PERM0, 256);
  GEMM(false, false, true,  false, false, false, 512, 256, 128,
       ws + OFF_XA, ARENA, 128, ws + OFF_WLT, 0, 128, ws + OFF_XWT, ARENA, 512, (const float*)nullptr);
  GEMM(false, false, false, true,  false, true,  512, 256, 512,
       ws + OFF_AN0, ARENA, 512, ws + OFF_XWT, ARENA, 512, ws + OFF_HNODE, ARENA, 256, blast);

  // ---- readout ----
  pool_kernel<<<eg(B_ * E_), 256, 0, stream>>>(ws);
  mlp_gemm_kernel<<<eg(B_ * E_), 256, 0, stream>>>(ws, OFF_HG, OFF_H1, P1w, P1b, E_, E_);
  bn_kernel<<<1, 256, 0, stream>>>(ws, OFF_H1, g1, be1, E_);
  mlp_gemm_kernel<<<eg(B_ * H_), 256, 0, stream>>>(ws, OFF_H1, OFF_H2, P2w, P2b, E_, H_);
  bn_kernel<<<1, 128, 0, stream>>>(ws, OFF_H2, g2, be2, H_);
  mlp_out_kernel<<<1, 64, 0, stream>>>(ws, P3w, P3b, (float*)d_out);
}
